// Attention_58282706206917
// MI455X (gfx1250) — matrix operand order repaired and checked by reading, not compiled, not run on hardware
//
#include <hip/hip_runtime.h>
#include <hip/hip_bf16.h>
#include <cstdint>
#include <cstddef>

typedef __bf16 bf16_t;
typedef __attribute__((ext_vector_type(16))) __bf16 v16bf;
typedef __attribute__((ext_vector_type(8)))  __bf16 v8bf;
typedef __attribute__((ext_vector_type(8)))  float  v8f;
typedef __attribute__((ext_vector_type(4)))  unsigned int u32x4;
typedef __attribute__((ext_vector_type(8)))  int i32x8;
typedef __attribute__((ext_vector_type(4)))  int i32x4;

#define B_    2
#define N_    4096
#define DIM_  1024
#define H_    16
#define HD_   64
#define RP_   256
#define MTOT  8192
#define BH_   (B_*H_)

// ---------------- TDM availability (device pass only) ----------------
#ifdef __HIP_DEVICE_COMPILE__
#if __has_builtin(__builtin_amdgcn_tensor_load_to_lds)
#define HAVE_TDM 1
#endif
#endif
#ifndef HAVE_TDM
#define HAVE_TDM 0
#endif

// ---------------- WMMA helpers ----------------
__device__ __forceinline__ v8f wmma_bf16(v16bf a, v16bf b, v8f c) {
  return __builtin_amdgcn_wmma_f32_16x16x32_bf16(false, a, false, b, (short)0, c,
                                                 false, false);
}

// A fragment 16x32 bf16 from row-major [.., ld] at (0,0) of tile.
// lane m = lane&15, k-half = lane>>4: v[0..7]=k(h*8+i), v[8..15]=k(16+h*8+i)
__device__ __forceinline__ v16bf load_a_frag(const bf16_t* base, int ld) {
  int lane = threadIdx.x & 31;
  int m = lane & 15, kh = lane >> 4;
  const bf16_t* r = base + m * ld + kh * 8;
  v8bf lo = *(const v8bf*)(r);
  v8bf hi = *(const v8bf*)(r + 16);
  return __builtin_shufflevector(lo, hi, 0,1,2,3,4,5,6,7,8,9,10,11,12,13,14,15);
}

// B fragment 32x16: source stored as B[n, k] row-major (we compute A*B^T).
// lane n = lane&15, k-half = lane>>4: v[0..7]=B[n, h*8+i], v[8..15]=B[n, 16+h*8+i]
__device__ __forceinline__ v16bf load_b_frag(const bf16_t* base, int ld) {
  int lane = threadIdx.x & 31;
  int n = lane & 15, kb = (lane >> 4) << 3;
  const bf16_t* r = base + n * ld + kb;
  v8bf lo = *(const v8bf*)(r);
  v8bf hi = *(const v8bf*)(r + 16);
  return __builtin_shufflevector(lo, hi, 0,1,2,3,4,5,6,7,8,9,10,11,12,13,14,15);
}

// ---------------- Tensor Data Mover: 2D tile -> LDS ----------------
#if HAVE_TDM
__device__ __forceinline__ void tdm_load_tile(unsigned lds_off, const void* gp,
                                              unsigned tdim0, unsigned tdim1,
                                              unsigned tiled0, unsigned tiled1,
                                              unsigned long long stride_elems) {
  unsigned long long ga = (unsigned long long)(uintptr_t)gp;
  u32x4 g0;
  g0[0] = 1u;                                   // count=1, user descriptor
  g0[1] = lds_off;                              // lds_addr
  g0[2] = (unsigned)ga;                         // global_addr[31:0]
  g0[3] = ((unsigned)(ga >> 32) & 0x01FFFFFFu) | 0x80000000u; // [56:32] | type=2
  i32x8 g1;
  // data_size=2B (code1), pad_enable, pad every 16 dwords (code3) by 4 dwords (code3)
  g1[0] = (1 << 16) | (1 << 20) | (3 << 22) | (3 << 25);
  g1[1] = (int)((tdim0 & 0xFFFFu) << 16);                        // tensor_dim0 lo
  g1[2] = (int)((tdim0 >> 16) & 0xFFFFu) | (int)((tdim1 & 0xFFFFu) << 16);
  g1[3] = (int)((tdim1 >> 16) & 0xFFFFu) | (int)(tiled0 << 16);  // tile_dim0
  g1[4] = (int)(tiled1 & 0xFFFFu);                               // tile_dim1
  g1[5] = (int)(stride_elems & 0xFFFFFFFFull);                   // dim0_stride lo
  g1[6] = (int)((stride_elems >> 32) & 0xFFFFull);               // dim0_stride hi
  g1[7] = 0;
  i32x4 z4 = {0, 0, 0, 0};
  i32x8 z8 = {0, 0, 0, 0, 0, 0, 0, 0};
  __builtin_amdgcn_tensor_load_to_lds(g0, g1, z4, z4, z8, 0);
}
#endif

// ---------------- fp32 -> bf16 convert ----------------
__global__ void k_cvt(const float* __restrict__ s, bf16_t* __restrict__ d, long n) {
  long i = (long)blockIdx.x * blockDim.x + threadIdx.x;
  long st = (long)gridDim.x * blockDim.x;
  for (; i < n; i += st) d[i] = (bf16_t)s[i];
}

// ---------------- big GEMM: C[M x Ncols] = A[M,K] * Bm[Ncols,K]^T ----------------
// EPI 0: QKV epilogue (Ncols=3072, scatter to head-split Q/K/V bf16, mask K/V)
// EPI 1: FF epilogue  (Ncols=1024, +bias, fp32 out)
template <int EPI>
__global__ __launch_bounds__(256) void k_gemm128(
    const bf16_t* __restrict__ A, const bf16_t* __restrict__ Bm, int K,
    const float* __restrict__ bq, const float* __restrict__ bk,
    const float* __restrict__ bv, const float* __restrict__ maskp,
    bf16_t* __restrict__ Qb, bf16_t* __restrict__ Kb, bf16_t* __restrict__ Vb,
    const float* __restrict__ bias_ff, float* __restrict__ outF) {
  __shared__ bf16_t As[128 * 40];
  __shared__ bf16_t Bs[128 * 40];
  const int tid = threadIdx.x;
  const int wid = tid >> 5, lane = tid & 31;
  const int wrow = wid >> 2, wcol = wid & 3;  // 2 x 4 waves, 64x32 wave tile
  const int cn = lane & 15, cm = (lane >> 4) << 3;
  const int row0 = blockIdx.x * 128, col0 = blockIdx.y * 128;

  v8f acc[4][2] = {};
  for (int k0 = 0; k0 < K; k0 += 32) {
#if HAVE_TDM
    if (tid < 32) {
      tdm_load_tile((unsigned)(uintptr_t)As, A + (size_t)row0 * K + k0,
                    (unsigned)K, 0x100000u, 32u, 128u, (unsigned long long)K);
      tdm_load_tile((unsigned)(uintptr_t)Bs, Bm + (size_t)col0 * K + k0,
                    (unsigned)K, 0x100000u, 32u, 128u, (unsigned long long)K);
      __builtin_amdgcn_s_wait_tensorcnt(0);
    }
#else
    for (int i = tid; i < 128 * 4; i += 256) {
      int r = i >> 2, c = (i & 3) * 8;
      *(v8bf*)(As + r * 40 + c) = *(const v8bf*)(A + (size_t)(row0 + r) * K + k0 + c);
    }
    for (int i = tid; i < 128 * 4; i += 256) {
      int r = i >> 2, c = (i & 3) * 8;
      *(v8bf*)(Bs + r * 40 + c) = *(const v8bf*)(Bm + (size_t)(col0 + r) * K + k0 + c);
    }
#endif
    __syncthreads();
    v16bf af[4], bfr[2];
#pragma unroll
    for (int i = 0; i < 4; ++i) af[i] = load_a_frag(As + (wrow * 64 + i * 16) * 40, 40);
#pragma unroll
    for (int j = 0; j < 2; ++j) bfr[j] = load_b_frag(Bs + (wcol * 32 + j * 16) * 40, 40);
#pragma unroll
    for (int i = 0; i < 4; ++i)
#pragma unroll
      for (int j = 0; j < 2; ++j) acc[i][j] = wmma_bf16(af[i], bfr[j], acc[i][j]);
    __syncthreads();
  }

#pragma unroll
  for (int i = 0; i < 4; ++i)
#pragma unroll
    for (int j = 0; j < 2; ++j)
#pragma unroll
      for (int e = 0; e < 8; ++e) {
        int gm = row0 + wrow * 64 + i * 16 + cm + e;
        int gn = col0 + wcol * 32 + j * 16 + cn;
        float v = acc[i][j][e];
        if (EPI == 0) {
          int sel = gn >> 10, dc = gn & 1023;
          int b = gm >> 12, ns = gm & 4095;
          int h = dc >> 6, d = dc & 63;
          size_t o = (((size_t)(b * H_ + h)) * N_ + ns) * HD_ + d;
          if (sel == 0)      Qb[o] = (bf16_t)(v + bq[dc]);
          else if (sel == 1) Kb[o] = (bf16_t)((v + bk[dc]) * maskp[b * N_ + ns]);
          else               Vb[o] = (bf16_t)((v + bv[dc]) * maskp[b * N_ + ns]);
        } else {
          outF[(size_t)gm * DIM_ + gn] = v + bias_ff[gn];
        }
      }
}

// ---------------- diag: sum(x^2)/128 per (b,h,n) over 64 elems ----------------
__global__ void k_diag(const bf16_t* __restrict__ s, float* __restrict__ o, int total) {
  int i = blockIdx.x * blockDim.x + threadIdx.x;
  if (i >= total) return;
  const bf16_t* p = s + (size_t)i * HD_;
  float a = 0.f;
#pragma unroll 8
  for (int j = 0; j < HD_; ++j) { float v = (float)p[j]; a += v * v; }
  o[i] = a * (1.0f / 128.0f);
}

// ---------------- feature map: dd = 0.125 * D @ proj^T, fused epilogue ----------
// ISQ=1: qp = 1/16 * (exp(dd - diag - rowmax) + eps) -> bf16
// ISQ=0: write dd bf16 + per-block max (global key max handled later)
template <int ISQ>
__global__ __launch_bounds__(256) void k_feat(
    const bf16_t* __restrict__ Dsrc, const bf16_t* __restrict__ Pb,
    const float* __restrict__ diag, bf16_t* __restrict__ outp,
    float* __restrict__ blockmax) {
  __shared__ float sdd[32 * 260];
  __shared__ float sred[32 * 8];
  __shared__ float srow[32];
  const int bh = blockIdx.y, nt = blockIdx.x;  // 32-row tile
  const int tid = threadIdx.x;
  const int wid = tid >> 5, lane = tid & 31;
  const int wm = wid >> 2, wn = wid & 3;  // 2 x 4 waves: 16 rows x 64 cols each
  const int cn = lane & 15, cm = (lane >> 4) << 3;
  const bf16_t* Ah = Dsrc + ((size_t)bh * N_ + nt * 32) * HD_;

  v8f acc[4] = {};
#pragma unroll
  for (int kt = 0; kt < 2; ++kt) {
    v16bf a = load_a_frag(Ah + (wm * 16) * HD_ + kt * 32, HD_);
#pragma unroll
    for (int j = 0; j < 4; ++j) {
      v16bf b = load_b_frag(Pb + (wn * 64 + j * 16) * HD_ + kt * 32, HD_);
      acc[j] = wmma_bf16(a, b, acc[j]);
    }
  }
#pragma unroll
  for (int j = 0; j < 4; ++j)
#pragma unroll
    for (int e = 0; e < 8; ++e)
      sdd[(wm * 16 + cm + e) * 260 + wn * 64 + j * 16 + cn] = 0.125f * acc[j][e];
  __syncthreads();

  { // row max (32 rows x 256 cols)
    int r = tid >> 3, part = tid & 7;
    float mx = -1e30f;
    for (int c = part * 32; c < part * 32 + 32; ++c) mx = fmaxf(mx, sdd[r * 260 + c]);
    sred[r * 8 + part] = mx;
  }
  __syncthreads();
  if (tid < 32) {
    float m2 = sred[tid * 8];
#pragma unroll
    for (int p = 1; p < 8; ++p) m2 = fmaxf(m2, sred[tid * 8 + p]);
    srow[tid] = m2;
    if (!ISQ) {
      float v = m2;
#pragma unroll
      for (int off = 16; off > 0; off >>= 1) v = fmaxf(v, __shfl_xor(v, off, 32));
      if (tid == 0) blockmax[bh * 128 + nt] = v;
    }
  }
  __syncthreads();

  for (int i = tid; i < 32 * 256; i += 256) {
    int r = i >> 8, c = i & 255;
    int gr = nt * 32 + r;
    float dd = sdd[r * 260 + c];
    float v;
    if (ISQ)
      v = 0.0625f * (__expf(dd - diag[(size_t)bh * N_ + gr] - srow[r]) + 1e-4f);
    else
      v = dd;  // exp applied after global max is known
    outp[((size_t)bh * N_ + gr) * RP_ + c] = (bf16_t)v;
  }
}

__global__ void k_kmax_reduce(const float* __restrict__ bm, float* __restrict__ km) {
  int bh = threadIdx.x;  // 32 threads
  float m = -1e30f;
  for (int i = 0; i < 128; ++i) m = fmaxf(m, bm[bh * 128 + i]);
  km[bh] = m;
}

__global__ void k_kexp(const bf16_t* __restrict__ kdd, const float* __restrict__ kdiag,
                       const float* __restrict__ kmax, bf16_t* __restrict__ kp) {
  size_t i = (size_t)blockIdx.x * 256 + threadIdx.x;  // total = BH*N*RP
  int bh = (int)(i >> 20);
  float v = 0.0625f * (__expf((float)kdd[i] - kdiag[i >> 8] - kmax[bh]) + 1e-4f);
  kp[i] = (bf16_t)v;
}

__global__ void k_ksum(const bf16_t* __restrict__ kp, float* __restrict__ ksum) {
  int bh = blockIdx.x, j = threadIdx.x;  // 256 threads
  const bf16_t* p = kp + (size_t)bh * N_ * RP_ + j;
  float s = 0.f;
  for (int n = 0; n < N_; ++n) s += (float)p[(size_t)n * RP_];
  ksum[bh * RP_ + j] = s;
}

__global__ __launch_bounds__(256) void k_dinv(const bf16_t* __restrict__ qp,
                                              const float* __restrict__ ksum,
                                              float* __restrict__ dinv) {
  int wid = threadIdx.x >> 5, lane = threadIdx.x & 31;
  int row = blockIdx.x * 8 + wid;  // 0..BH*N-1
  int bh = row >> 12;
  const bf16_t* p = qp + (size_t)row * RP_ + lane * 8;
  const float* k = ksum + bh * RP_ + lane * 8;
  float s = 0.f;
#pragma unroll
  for (int i = 0; i < 8; ++i) s += (float)p[i] * k[i];
#pragma unroll
  for (int off = 16; off > 0; off >>= 1) s += __shfl_xor(s, off, 32);
  if (lane == 0) dinv[row] = 1.0f / s;
}

// ---------------- context: ctxT[bh][e][r] = sum_n kp[n,r]*Vm[n,e] ----------------
__global__ __launch_bounds__(256) void k_ctx(const bf16_t* __restrict__ kp,
                                             const bf16_t* __restrict__ Vb,
                                             bf16_t* __restrict__ ctxT) {
  __shared__ bf16_t skp[256 * 40];  // kp^T tile: [r][n]
  __shared__ bf16_t svt[64 * 40];   // V^T tile:  [e][n]
  const int bh = blockIdx.x;
  const int tid = threadIdx.x;
  const int wid = tid >> 5, lane = tid & 31;
  const int wr = wid >> 1, we = wid & 1;  // 4 x 2 waves, 64r x 32e wave tiles
  const int cn = lane & 15, cm = (lane >> 4) << 3;

  v8f acc[4][2] = {};
  for (int n0 = 0; n0 < N_; n0 += 32) {
    for (int i = tid; i < 32 * 256; i += 256) {
      int nl = i >> 8, r = i & 255;
      skp[r * 40 + nl] = kp[((size_t)bh * N_ + n0 + nl) * RP_ + r];
    }
    for (int i = tid; i < 32 * 64; i += 256) {
      int nl = i >> 6, e = i & 63;
      svt[e * 40 + nl] = Vb[((size_t)bh * N_ + n0 + nl) * HD_ + e];
    }
    __syncthreads();
    v16bf a[4], b[2];
#pragma unroll
    for (int i = 0; i < 4; ++i) a[i] = load_a_frag(skp + (wr * 64 + i * 16) * 40, 40);
#pragma unroll
    for (int j = 0; j < 2; ++j) b[j] = load_b_frag(svt + (we * 32 + j * 16) * 40, 40);
#pragma unroll
    for (int i = 0; i < 4; ++i)
#pragma unroll
      for (int j = 0; j < 2; ++j) acc[i][j] = wmma_bf16(a[i], b[j], acc[i][j]);
    __syncthreads();
  }
#pragma unroll
  for (int i = 0; i < 4; ++i)
#pragma unroll
    for (int j = 0; j < 2; ++j)
#pragma unroll
      for (int e = 0; e < 8; ++e) {
        int r = wr * 64 + i * 16 + cm + e;
        int ec = we * 32 + j * 16 + cn;
        ctxT[(size_t)bh * HD_ * RP_ + ec * RP_ + r] = (bf16_t)acc[i][j][e];
      }
}

// ---------------- attn: out = D_inv * (qp @ ctx) -> [b,n,h*64+e] bf16 ----------
__global__ __launch_bounds__(256) void k_attn(const bf16_t* __restrict__ qp,
                                              const bf16_t* __restrict__ ctxT,
                                              const float* __restrict__ dinv,
                                              bf16_t* __restrict__ attn2) {
  const int bh = blockIdx.y, nt = blockIdx.x;  // 128-row tile
  const int tid = threadIdx.x;
  const int wid = tid >> 5, lane = tid & 31;
  const int wm = wid >> 1, wn = wid & 1;  // 4 x 2 waves, 32m x 32e
  const int cn = lane & 15, cm = (lane >> 4) << 3;
  const bf16_t* Aq = qp + ((size_t)bh * N_ + nt * 128) * RP_;
  const bf16_t* Bc = ctxT + (size_t)bh * HD_ * RP_;

  v8f acc[2][2] = {};
#pragma unroll
  for (int kt = 0; kt < 8; ++kt) {
    v16bf a[2], b[2];
#pragma unroll
    for (int i = 0; i < 2; ++i) a[i] = load_a_frag(Aq + (wm * 32 + i * 16) * RP_ + kt * 32, RP_);
#pragma unroll
    for (int j = 0; j < 2; ++j) b[j] = load_b_frag(Bc + (wn * 32 + j * 16) * RP_ + kt * 32, RP_);
#pragma unroll
    for (int i = 0; i < 2; ++i)
#pragma unroll
      for (int j = 0; j < 2; ++j) acc[i][j] = wmma_bf16(a[i], b[j], acc[i][j]);
  }
  int b = bh >> 4, h = bh & 15;
#pragma unroll
  for (int i = 0; i < 2; ++i)
#pragma unroll
    for (int j = 0; j < 2; ++j)
#pragma unroll
      for (int e = 0; e < 8; ++e) {
        int nr = nt * 128 + wm * 32 + i * 16 + cm + e;
        int ec = wn * 32 + j * 16 + cn;
        float v = acc[i][j][e] * dinv[(size_t)bh * N_ + nr];
        attn2[((size_t)b * N_ + nr) * DIM_ + h * HD_ + ec] = (bf16_t)v;
      }
}

// ---------------- launch ----------------
extern "C" void kernel_launch(void* const* d_in, const int* in_sizes, int n_in,
                              void* d_out, int out_size, void* d_ws, size_t ws_size,
                              hipStream_t stream) {
  (void)in_sizes; (void)n_in; (void)out_size; (void)ws_size;
  const float* X    = (const float*)d_in[0];
  const float* mask = (const float*)d_in[1];
  const float* Wq   = (const float*)d_in[2];
  const float* bq   = (const float*)d_in[3];
  const float* Wk   = (const float*)d_in[4];
  const float* bk   = (const float*)d_in[5];
  const float* Wv   = (const float*)d_in[6];
  const float* bv   = (const float*)d_in[7];
  const float* Wff  = (const float*)d_in[8];
  const float* bff  = (const float*)d_in[9];
  const float* proj = (const float*)d_in[10];
  float* out = (float*)d_out;

  char* ws = (char*)d_ws;
  size_t off = 0;
  auto alloc = [&](size_t bytes) {
    size_t o = off;
    off += (bytes + 255) & ~(size_t)255;
    return o;
  };
  bf16_t* Xb    = (bf16_t*)(ws + alloc((size_t)MTOT * DIM_ * 2));
  bf16_t* Wqkvb = (bf16_t*)(ws + alloc((size_t)3 * DIM_ * DIM_ * 2));
  bf16_t* Wffb  = (bf16_t*)(ws + alloc((size_t)DIM_ * DIM_ * 2));
  bf16_t* Pb    = (bf16_t*)(ws + alloc((size_t)RP_ * HD_ * 2));
  bf16_t* Qb    = (bf16_t*)(ws + alloc((size_t)BH_ * N_ * HD_ * 2));
  bf16_t* Kb    = (bf16_t*)(ws + alloc((size_t)BH_ * N_ * HD_ * 2));
  bf16_t* Vb    = (bf16_t*)(ws + alloc((size_t)BH_ * N_ * HD_ * 2));
  float*  qdiag = (float*)(ws + alloc((size_t)BH_ * N_ * 4));
  float*  kdiag = (float*)(ws + alloc((size_t)BH_ * N_ * 4));
  bf16_t* qp    = (bf16_t*)(ws + alloc((size_t)BH_ * N_ * RP_ * 2));
  bf16_t* kdd   = (bf16_t*)(ws + alloc((size_t)BH_ * N_ * RP_ * 2));
  bf16_t* kp    = (bf16_t*)(ws + alloc((size_t)BH_ * N_ * RP_ * 2));
  float*  kmaxb = (float*)(ws + alloc((size_t)BH_ * 128 * 4));
  float*  kmax  = (float*)(ws + alloc((size_t)BH_ * 4));
  float*  ksum  = (float*)(ws + alloc((size_t)BH_ * RP_ * 4));
  float*  dinv  = (float*)(ws + alloc((size_t)BH_ * N_ * 4));
  bf16_t* ctxT  = (bf16_t*)(ws + alloc((size_t)BH_ * HD_ * RP_ * 2));
  bf16_t* attn2 = (bf16_t*)(ws + alloc((size_t)MTOT * DIM_ * 2));

  // 1) fp32 -> bf16 packs
  k_cvt<<<2048, 256, 0, stream>>>(X, Xb, (long)MTOT * DIM_);
  k_cvt<<<512, 256, 0, stream>>>(Wq, Wqkvb, (long)DIM_ * DIM_);
  k_cvt<<<512, 256, 0, stream>>>(Wk, Wqkvb + (size_t)DIM_ * DIM_, (long)DIM_ * DIM_);
  k_cvt<<<512, 256, 0, stream>>>(Wv, Wqkvb + (size_t)2 * DIM_ * DIM_, (long)DIM_ * DIM_);
  k_cvt<<<512, 256, 0, stream>>>(Wff, Wffb, (long)DIM_ * DIM_);
  k_cvt<<<64, 256, 0, stream>>>(proj, Pb, (long)RP_ * HD_);

  // 2) fused QKV GEMM (8192 x 3072 x 1024) -> head-split bf16 Q, K*m, V*m
  k_gemm128<0><<<dim3(MTOT / 128, 3072 / 128), 256, 0, stream>>>(
      Xb, Wqkvb, DIM_, bq, bk, bv, mask, Qb, Kb, Vb, nullptr, nullptr);

  // 3) diag terms
  k_diag<<<(BH_ * N_) / 256, 256, 0, stream>>>(Qb, qdiag, BH_ * N_);
  k_diag<<<(BH_ * N_) / 256, 256, 0, stream>>>(Kb, kdiag, BH_ * N_);

  // 4) feature maps
  k_feat<1><<<dim3(N_ / 32, BH_), 256, 0, stream>>>(Qb, Pb, qdiag, qp, nullptr);
  k_feat<0><<<dim3(N_ / 32, BH_), 256, 0, stream>>>(Kb, Pb, kdiag, kdd, kmaxb);
  k_kmax_reduce<<<1, BH_, 0, stream>>>(kmaxb, kmax);
  k_kexp<<<(BH_ * N_ * RP_) / 256, 256, 0, stream>>>(kdd, kdiag, kmax, kp);

  // 5) k_sum, D_inv
  k_ksum<<<BH_, RP_, 0, stream>>>(kp, ksum);
  k_dinv<<<(BH_ * N_) / 8, 256, 0, stream>>>(qp, ksum, dinv);

  // 6) context, attn
  k_ctx<<<BH_, 256, 0, stream>>>(kp, Vb, ctxT);
  k_attn<<<dim3(N_ / 128, BH_), 256, 0, stream>>>(qp, ctxT, dinv, attn2);

  // 7) output projection (8192 x 1024 x 1024), fp32 out
  k_gemm128<1><<<dim3(MTOT / 128, DIM_ / 128), 256, 0, stream>>>(
      attn2, Wffb, DIM_, nullptr, nullptr, nullptr, nullptr, nullptr, nullptr,
      nullptr, bff, out);
}
